// GCN_35673998360640
// MI455X (gfx1250) — matrix operand order repaired and checked by reading, not compiled, not run on hardware
//
#include <hip/hip_runtime.h>

typedef unsigned int   u32;
typedef unsigned short u16;
typedef __attribute__((ext_vector_type(16))) __bf16 v16bf;
typedef __attribute__((ext_vector_type(8)))  float  v8f;

#define B_   64
#define N_   1024
#define HID_ 64
#define OUT_ 32

__device__ __forceinline__ u16 f2bf(float f) {
  union { float f; u32 u; } x; x.f = f;
  u32 r = x.u + 0x7FFFu + ((x.u >> 16) & 1u);
  return (u16)(r >> 16);
}
__device__ __forceinline__ float bf2f(u16 h) {
  union { u32 u; float f; } x; x.u = ((u32)h) << 16;
  return x.f;
}
__device__ __forceinline__ u32 pack2(u16 lo, u16 hi) {
  return (u32)lo | ((u32)hi << 16);
}

// ---------------- Kernel 1: inverse degree (row sums of graph + self loop) ------------
// one wave32 per row
__global__ void __launch_bounds__(256) k_degree(const float* __restrict__ graph,
                                                float* __restrict__ inv_deg) {
  int wave = threadIdx.x >> 5;
  int lane = threadIdx.x & 31;
  int row  = blockIdx.x * 8 + wave;                 // [0, B_*N_)
  const float* gp = graph + (size_t)row * N_;
  float s = 0.f;
#pragma unroll
  for (int i = 0; i < 32; ++i) s += gp[lane + 32 * i];
#pragma unroll
  for (int m = 16; m >= 1; m >>= 1) s += __shfl_xor(s, m, 32);
  s += 1.0f;                                        // self loop
  if (lane == 0) inv_deg[row] = (s == 0.f) ? 0.f : 1.f / s;
}

// ---------------- Kernel 2: normalize + bf16 + A-fragment swizzle ---------------------
// A fragment (16x32 bf16): lane L holds row M=L%16; K list = L<16 ? {0..7,16..23}
//                                                             : {8..15,24..31}
// stored per tile as [lane][8 dwords] contiguous (1024B per tile), tiles ordered
// [b][mt(64)][kb(32)].
__global__ void __launch_bounds__(256) k_swizzle(const float* __restrict__ graph,
                                                 const float* __restrict__ inv_deg,
                                                 u32* __restrict__ g_swz) {
  int bIdx = blockIdx.x;                 // b*64 + mt
  int b  = bIdx >> 6;
  int mt = bIdx & 63;
  int wave = threadIdx.x >> 5;
  int lane = threadIdx.x & 31;
  int M   = lane & 15;
  int row = mt * 16 + M;
  float inv = inv_deg[b * N_ + row];
  const float* rowp = graph + (size_t)(b * N_ + row) * N_;
  int k0 = (lane >> 4) * 8;              // 0 or 8

  for (int kb = wave; kb < 32; kb += 8) {
    const float* p = rowp + kb * 32 + k0;
    float4 f0 = *(const float4*)(p);
    float4 f1 = *(const float4*)(p + 4);
    float4 f2 = *(const float4*)(p + 16);
    float4 f3 = *(const float4*)(p + 20);
    float v[16] = { f0.x, f0.y, f0.z, f0.w, f1.x, f1.y, f1.z, f1.w,
                    f2.x, f2.y, f2.z, f2.w, f3.x, f3.y, f3.z, f3.w };
    int kA = kb * 32 + k0;
#pragma unroll
    for (int j = 0; j < 16; ++j) {
      int k = kA + (j < 8 ? j : j + 8);
      float val = v[j] + ((k == row) ? 1.0f : 0.0f);
      v[j] = val * inv;
    }
    u32 d[8];
#pragma unroll
    for (int q = 0; q < 8; ++q) d[q] = pack2(f2bf(v[2 * q]), f2bf(v[2 * q + 1]));
    u32* dst = g_swz + ((size_t)bIdx * 32 + kb) * 256 + lane * 8;
    *(uint4*)(dst)     = make_uint4(d[0], d[1], d[2], d[3]);
    *(uint4*)(dst + 4) = make_uint4(d[4], d[5], d[6], d[7]);
  }
}

// ---------------- Kernel 3: linear1, output in B-fragment swizzled layout -------------
// B fragment (32x16 bf16): lane = n + 16*((k/8)%2), vgpr v holds k = 8*(lane/16) + 16*(v/4) + 2*(v%4) (+1).
// dword index d decodes: blockFlat = b*128 + kb*4 + nb ; r = d%256 ; lane=r/8 ; v=r%8.
__global__ void __launch_bounds__(256) k_linear1(const float* __restrict__ x,
                                                 const float* __restrict__ W1,
                                                 const float* __restrict__ b1,
                                                 u32* __restrict__ h0_swz) {
  size_t d = (size_t)blockIdx.x * 256 + threadIdx.x;   // total 2,097,152 dwords
  u32 r = (u32)(d & 255);
  size_t bf = d >> 8;
  int nb = (int)(bf & 3);
  int kb = (int)((bf >> 2) & 31);
  int b  = (int)(bf >> 7);
  int lane = r >> 3, v = r & 7;
  int c = nb * 16 + (lane & 15);
  int m = kb * 32 + (lane >> 4) * 8 + (v & 3) * 2 + (v >> 2) * 16;
  const float* xr = x + (size_t)(b * N_ + m) * 16;
  const float* w  = W1 + c * 16;
  float s0 = b1[c], s1 = s0;
#pragma unroll
  for (int i = 0; i < 16; ++i) {
    float wi = w[i];
    s0 += xr[i] * wi;
    s1 += xr[16 + i] * wi;
  }
  h0_swz[d] = pack2(f2bf(s0), f2bf(s1));
}

// ---------------- Kernel 4: propagation 1 (g @ h0), WMMA bf16, ReLU -------------------
// block: 256 thr (8 waves), 128KB dynamic LDS = whole h0 slab of one batch.
// A/B fragments software double-buffered so WMMAs never wait on same-iteration loads.
__global__ void __launch_bounds__(256) k_prop1(const u32* __restrict__ g_swz,
                                               const u32* __restrict__ h0_swz,
                                               u16* __restrict__ h1) {
  extern __shared__ u32 smem[];
  int b  = blockIdx.x >> 3;
  int rg = blockIdx.x & 7;
  {
    const uint4* src = (const uint4*)(h0_swz + (size_t)b * 32768);
    uint4* dst = (uint4*)smem;
    for (int i = threadIdx.x; i < 8192; i += 256) dst[i] = src[i];
  }
  __syncthreads();

  int wave = threadIdx.x >> 5;
  int lane = threadIdx.x & 31;
  int mt = rg * 8 + wave;

  v8f z = {0.f, 0.f, 0.f, 0.f, 0.f, 0.f, 0.f, 0.f};
  v8f acc0 = z, acc1 = z, acc2 = z, acc3 = z;

  const u32* gA = g_swz + (size_t)(b * 64 + mt) * 32 * 256 + lane * 8;
  const u32* bB = smem + lane * 8;

  v16bf a  = *(const v16bf*)(gA);
  v16bf b0 = *(const v16bf*)(bB);
  v16bf b1v = *(const v16bf*)(bB + 256);
  v16bf b2v = *(const v16bf*)(bB + 512);
  v16bf b3v = *(const v16bf*)(bB + 768);

  for (int kb = 0; kb < 32; ++kb) {
    int kn = (kb + 1) & 31;              // wrap: last iter reloads block 0 (discarded)
    // prefetch the A-stream ahead of the XDL pipe (global_prefetch_b8)
    __builtin_prefetch((const void*)(gA + ((kb + 8) & 31) * 256), 0, 1);
    v16bf aN  = *(const v16bf*)(gA + kn * 256);
    const u32* bn = bB + kn * 1024;
    v16bf c0 = *(const v16bf*)(bn);
    v16bf c1 = *(const v16bf*)(bn + 256);
    v16bf c2 = *(const v16bf*)(bn + 512);
    v16bf c3 = *(const v16bf*)(bn + 768);
    acc0 = __builtin_amdgcn_wmma_f32_16x16x32_bf16(false, a, false, b0,  (short)0, acc0, false, false);
    acc1 = __builtin_amdgcn_wmma_f32_16x16x32_bf16(false, a, false, b1v, (short)0, acc1, false, false);
    acc2 = __builtin_amdgcn_wmma_f32_16x16x32_bf16(false, a, false, b2v, (short)0, acc2, false, false);
    acc3 = __builtin_amdgcn_wmma_f32_16x16x32_bf16(false, a, false, b3v, (short)0, acc3, false, false);
    a = aN; b0 = c0; b1v = c1; b2v = c2; b3v = c3;
  }

  // C layout: lanes 0-15 -> n=lane, M=r ; lanes 16-31 -> n=lane-16, M=r+8
  int n  = lane & 15;
  int hi = lane >> 4;
  v8f accs[4] = {acc0, acc1, acc2, acc3};
#pragma unroll
  for (int nb = 0; nb < 4; ++nb) {
#pragma unroll
    for (int r = 0; r < 8; ++r) {
      int m = mt * 16 + r + 8 * hi;
      int c = nb * 16 + n;
      float vv = accs[nb][r];
      vv = vv > 0.f ? vv : 0.f;                        // ReLU
      h1[(size_t)(b * N_ + m) * HID_ + c] = f2bf(vv);
    }
  }
}

// ---------------- Kernel 5: linear2, output in B-fragment swizzled layout -------------
__global__ void __launch_bounds__(256) k_linear2(const u16* __restrict__ h1,
                                                 const float* __restrict__ W2,
                                                 const float* __restrict__ b2,
                                                 u32* __restrict__ o0_swz) {
  size_t d = (size_t)blockIdx.x * 256 + threadIdx.x;   // total 1,048,576 dwords
  u32 r = (u32)(d & 255);
  size_t bf = d >> 8;
  int nb = (int)(bf & 1);
  int kb = (int)((bf >> 1) & 31);
  int b  = (int)(bf >> 6);
  int lane = r >> 3, v = r & 7;
  int c = nb * 16 + (lane & 15);
  int m = kb * 32 + (lane >> 4) * 8 + (v & 3) * 2 + (v >> 2) * 16;
  const u32* h0p = (const u32*)(h1 + (size_t)(b * N_ + m) * HID_);        // row m
  const u32* h1p = (const u32*)(h1 + (size_t)(b * N_ + m + 1) * HID_);    // row m+1
  const float* w = W2 + c * HID_;
  float s0 = b2[c], s1 = s0;
#pragma unroll
  for (int i = 0; i < 32; ++i) {
    u32 q0 = h0p[i], q1 = h1p[i];
    float w0 = w[2 * i], w1 = w[2 * i + 1];
    s0 += bf2f((u16)q0) * w0 + bf2f((u16)(q0 >> 16)) * w1;
    s1 += bf2f((u16)q1) * w0 + bf2f((u16)(q1 >> 16)) * w1;
  }
  o0_swz[d] = pack2(f2bf(s0), f2bf(s1));
}

// ---------------- Kernel 6: propagation 2 (g @ o0), WMMA bf16, ReLU, fp32 out ---------
__global__ void __launch_bounds__(256) k_prop2(const u32* __restrict__ g_swz,
                                               const u32* __restrict__ o0_swz,
                                               float* __restrict__ out) {
  extern __shared__ u32 smem[];
  int b  = blockIdx.x >> 3;
  int rg = blockIdx.x & 7;
  {
    const uint4* src = (const uint4*)(o0_swz + (size_t)b * 16384);
    uint4* dst = (uint4*)smem;
    for (int i = threadIdx.x; i < 4096; i += 256) dst[i] = src[i];
  }
  __syncthreads();

  int wave = threadIdx.x >> 5;
  int lane = threadIdx.x & 31;
  int mt = rg * 8 + wave;

  v8f z = {0.f, 0.f, 0.f, 0.f, 0.f, 0.f, 0.f, 0.f};
  v8f acc0 = z, acc1 = z;

  const u32* gA = g_swz + (size_t)(b * 64 + mt) * 32 * 256 + lane * 8;
  const u32* bB = smem + lane * 8;

  v16bf a   = *(const v16bf*)(gA);
  v16bf b0  = *(const v16bf*)(bB);
  v16bf b1v = *(const v16bf*)(bB + 256);

  for (int kb = 0; kb < 32; ++kb) {
    int kn = (kb + 1) & 31;
    __builtin_prefetch((const void*)(gA + ((kb + 8) & 31) * 256), 0, 1);
    v16bf aN = *(const v16bf*)(gA + kn * 256);
    const u32* bn = bB + kn * 512;
    v16bf c0 = *(const v16bf*)(bn);
    v16bf c1 = *(const v16bf*)(bn + 256);
    acc0 = __builtin_amdgcn_wmma_f32_16x16x32_bf16(false, a, false, b0,  (short)0, acc0, false, false);
    acc1 = __builtin_amdgcn_wmma_f32_16x16x32_bf16(false, a, false, b1v, (short)0, acc1, false, false);
    a = aN; b0 = c0; b1v = c1;
  }

  int n  = lane & 15;
  int hi = lane >> 4;
  v8f accs[2] = {acc0, acc1};
#pragma unroll
  for (int nb = 0; nb < 2; ++nb) {
#pragma unroll
    for (int r = 0; r < 8; ++r) {
      int m = mt * 16 + r + 8 * hi;
      int c = nb * 16 + n;
      float vv = accs[nb][r];
      out[(size_t)(b * N_ + m) * OUT_ + c] = vv > 0.f ? vv : 0.f;
    }
  }
}

extern "C" void kernel_launch(void* const* d_in, const int* in_sizes, int n_in,
                              void* d_out, int out_size, void* d_ws, size_t ws_size,
                              hipStream_t stream) {
  const float* x     = (const float*)d_in[0];
  const float* graph = (const float*)d_in[1];
  const float* W1    = (const float*)d_in[2];
  const float* b1    = (const float*)d_in[3];
  const float* W2    = (const float*)d_in[4];
  const float* b2    = (const float*)d_in[5];
  float* out = (float*)d_out;

  char* ws = (char*)d_ws;
  const size_t G_SWZ_BYTES  = (size_t)B_ * N_ * N_ * 2;        // 128 MB
  const size_t INVDEG_BYTES = (size_t)B_ * N_ * 4;             // 256 KB
  const size_t H0_BYTES     = (size_t)B_ * N_ * HID_ * 2;      // 8 MB
  const size_t H1_BYTES     = (size_t)B_ * N_ * HID_ * 2;      // 8 MB

  u32*   g_swz   = (u32*)ws;
  float* inv_deg = (float*)(ws + G_SWZ_BYTES);
  u32*   h0_swz  = (u32*)(ws + G_SWZ_BYTES + INVDEG_BYTES);
  u16*   h1      = (u16*)(ws + G_SWZ_BYTES + INVDEG_BYTES + H0_BYTES);
  u32*   o0_swz  = (u32*)(ws + G_SWZ_BYTES + INVDEG_BYTES + H0_BYTES + H1_BYTES);

  hipFuncSetAttribute((const void*)k_prop1, hipFuncAttributeMaxDynamicSharedMemorySize, 131072);
  hipFuncSetAttribute((const void*)k_prop2, hipFuncAttributeMaxDynamicSharedMemorySize, 65536);

  k_degree <<<(B_ * N_) / 8, 256, 0, stream>>>(graph, inv_deg);
  k_swizzle<<<B_ * 64,       256, 0, stream>>>(graph, inv_deg, g_swz);
  k_linear1<<<8192,          256, 0, stream>>>(x, W1, b1, h0_swz);
  k_prop1  <<<B_ * 8,        256, 131072, stream>>>(g_swz, h0_swz, h1);
  k_linear2<<<4096,          256, 0, stream>>>(h1, W2, b2, o0_swz);
  k_prop2  <<<B_ * 8,        256, 65536, stream>>>(g_swz, o0_swz, out);
}
